// MoEGPTOSS_25958782337094
// MI455X (gfx1250) — matrix operand order repaired and checked by reading, not compiled, not run on hardware
//
#include <hip/hip_runtime.h>

#define HDIM   2048
#define NTOK   8192          // B*S = 4*2048
#define NEXP   8
#define CAP    8224          // 8192 + one tile of padding
#define TILE_M 32
#define TILE_N 256
#define TILE_K 32
#define KSTEPS (HDIM / TILE_K)   // 64
#define GRID_N (HDIM / TILE_N)   // 8
#define MAX_TILES 264            // ceil(8192/32) + NEXP padding tiles

// padded LDS/ws tile row: 32 k-halves + 8 pad halves = 80 bytes (16B aligned)
#define KROW   40
#define W_TILE_BYTES ((size_t)TILE_N * KROW * 2)            // 20480
#define W_TILES      ((size_t)NEXP * KSTEPS * GRID_N)       // 4096
#define WPART_BYTES  (W_TILES * W_TILE_BYTES)               // 80 MB
#define HPART_BYTES  ((size_t)NTOK * HDIM * 2)              // 32 MB
#define META_BYTES   ((size_t)1 << 20)
#define REQ_BYTES    (META_BYTES + 2 * WPART_BYTES + 2 * HPART_BYTES)

typedef __attribute__((ext_vector_type(16))) __bf16         v16bf;
typedef __attribute__((ext_vector_type(8)))  float          v8f;
typedef __attribute__((ext_vector_type(4)))  unsigned int   u32x4;
typedef __attribute__((ext_vector_type(4)))  float          f32x4;
typedef __attribute__((ext_vector_type(4)))  unsigned short u16x4;
typedef __attribute__((ext_vector_type(8)))  int            i32x8;
typedef __attribute__((ext_vector_type(4)))  int            i32x4;

union FragBF {
    u32x4 u[2];
    v16bf v;
};

__device__ __forceinline__ unsigned short bf_hi(float x) {
    unsigned int u = __float_as_uint(x);
    u += 0x7FFFu + ((u >> 16) & 1u);           // round-to-nearest-even to bf16
    return (unsigned short)(u >> 16);
}
__device__ __forceinline__ float bf_val(unsigned short h) {
    return __uint_as_float(((unsigned int)h) << 16);
}
__device__ __forceinline__ void split_bf(float x, unsigned short& hi, unsigned short& lo) {
    hi = bf_hi(x);
    lo = bf_hi(x - bf_val(hi));
}

__device__ __forceinline__ v8f wmma_bf16(v16bf a, v16bf b, v8f c) {
    return __builtin_amdgcn_wmma_f32_16x16x32_bf16(false, a, false, b, (short)0, c,
                                                   false, false);
}

__device__ __forceinline__ int rfl(int x) { return __builtin_amdgcn_readfirstlane(x); }

// ---------------------------------------------------------------------------
// TDM descriptor helpers (CDNA5 ISA ch. 8 bitfields).
// ---------------------------------------------------------------------------
__device__ __forceinline__ i32x8 tdm_group1(unsigned data_size, unsigned pad_en,
                                            unsigned pad_interval, unsigned pad_amount,
                                            unsigned tdim0, unsigned tdim1,
                                            unsigned tile0, unsigned tile1,
                                            unsigned long long stride0) {
    i32x8 g;
    g[0] = rfl((int)((data_size << 16) | (pad_en << 20) | (pad_interval << 22) |
                     (pad_amount << 25)));
    g[1] = rfl((int)((tdim0 & 0xFFFFu) << 16));                       // [63:48] tdim0.lo
    g[2] = rfl((int)(((tdim0 >> 16) & 0xFFFFu) | ((tdim1 & 0xFFFFu) << 16)));
    g[3] = rfl((int)(((tdim1 >> 16) & 0xFFFFu) | (tile0 << 16)));
    g[4] = rfl((int)(tile1 & 0xFFFFu));                               // tile_dim2 = 0
    g[5] = rfl((int)(unsigned)(stride0 & 0xFFFFFFFFull));
    g[6] = rfl((int)(unsigned)((stride0 >> 32) & 0xFFFFull));         // stride1 = 0
    g[7] = 0;
    return g;
}

__device__ __forceinline__ void tdm_load(unsigned gather, unsigned lds_addr,
                                         unsigned long long ga,
                                         i32x8 g1, i32x4 g2, i32x4 g3) {
    u32x4 g0;
    g0[0] = (unsigned)rfl((int)(1u | (gather << 31)));                // count=1, 16b idx
    g0[1] = (unsigned)rfl((int)lds_addr);
    g0[2] = (unsigned)rfl((int)(unsigned)(ga & 0xFFFFFFFFull));
    g0[3] = (unsigned)rfl((int)(((unsigned)(ga >> 32) & 0x1FFFFFFu) | (2u << 30)));
    const i32x8 gx = {0, 0, 0, 0, 0, 0, 0, 0};                        // unused group
    __builtin_amdgcn_tensor_load_to_lds(g0, g1, g2, g3, gx, 0);       // 6-arg toolchain
}

// ---------------------------------------------------------------------------
// Kernel 0: zero the 16 per-expert counters (cnt0[8], cnt1[8] contiguous).
// ---------------------------------------------------------------------------
__global__ void moe_init_kernel(int* __restrict__ cnt) {
    if (threadIdx.x < 16) cnt[threadIdx.x] = 0;
}

// ---------------------------------------------------------------------------
// Kernel 1: gating (logits -> softmax -> top2 -> routing lists).
// ---------------------------------------------------------------------------
__global__ void __launch_bounds__(256)
moe_gate_kernel(const float* __restrict__ hidden,
                const float* __restrict__ gate_w,   // [H][E]
                const float* __restrict__ gate_b,   // [E]
                int* __restrict__ cnt0, int* __restrict__ cnt1,
                int* __restrict__ list0, int* __restrict__ list1,
                float* __restrict__ w0, float* __restrict__ w1) {
    __shared__ float sred[256];
    __shared__ float slog[NEXP];

    const int tok = blockIdx.x;
    const int tid = threadIdx.x;
    const float* hrow = hidden + (size_t)tok * HDIM;

    float acc[NEXP];
#pragma unroll
    for (int e = 0; e < NEXP; ++e) acc[e] = 0.0f;

    for (int h = tid; h < HDIM; h += 256) {
        const float x = hrow[h];
        const f32x4 g0 = *(const f32x4*)(gate_w + (size_t)h * NEXP);
        const f32x4 g1 = *(const f32x4*)(gate_w + (size_t)h * NEXP + 4);
        acc[0] += x * g0.x; acc[1] += x * g0.y; acc[2] += x * g0.z; acc[3] += x * g0.w;
        acc[4] += x * g1.x; acc[5] += x * g1.y; acc[6] += x * g1.z; acc[7] += x * g1.w;
    }

#pragma unroll
    for (int e = 0; e < NEXP; ++e) {
        sred[tid] = acc[e];
        __syncthreads();
        for (int s = 128; s > 0; s >>= 1) {
            if (tid < s) sred[tid] += sred[tid + s];
            __syncthreads();
        }
        if (tid == 0) slog[e] = sred[0] + gate_b[e];
        __syncthreads();
    }

    if (tid == 0) {
        float p[NEXP];
        float mx = slog[0];
#pragma unroll
        for (int e = 1; e < NEXP; ++e) mx = fmaxf(mx, slog[e]);
        float sum = 0.0f;
#pragma unroll
        for (int e = 0; e < NEXP; ++e) { p[e] = expf(slog[e] - mx); sum += p[e]; }
        const float inv = 1.0f / sum;
#pragma unroll
        for (int e = 0; e < NEXP; ++e) p[e] *= inv;

        int i0 = 0;
#pragma unroll
        for (int e = 1; e < NEXP; ++e) if (p[e] > p[i0]) i0 = e;   // ties -> lowest index
        int i1 = (i0 == 0) ? 1 : 0;
#pragma unroll
        for (int e = 0; e < NEXP; ++e) if (e != i0 && p[e] > p[i1]) i1 = e;

        w0[tok] = p[i0];
        w1[tok] = p[i1];
        const int pos0 = atomicAdd(&cnt0[i0], 1);
        list0[i0 * CAP + pos0] = tok;
        const int pos1 = atomicAdd(&cnt1[i1], 1);
        list1[i1 * CAP + pos1] = tok;
    }
}

// ---------------------------------------------------------------------------
// Kernel 2: pad lists with -1 sentinels; build tile prefix offsets.
// ---------------------------------------------------------------------------
__global__ void __launch_bounds__(64)
moe_finalize_kernel(const int* __restrict__ cnt0, const int* __restrict__ cnt1,
                    int* __restrict__ list0, int* __restrict__ list1,
                    int* __restrict__ ts0, int* __restrict__ ts1) {
    const int tid = threadIdx.x;
    for (int set = 0; set < 2; ++set) {
        const int* cnt = set ? cnt1 : cnt0;
        int* lst = set ? list1 : list0;
        int* ts  = set ? ts1 : ts0;
        for (int e = 0; e < NEXP; ++e) {
            const int c  = cnt[e];
            const int pc = (c + TILE_M - 1) & ~(TILE_M - 1);
            for (int i = c + tid; i < pc; i += 64) lst[e * CAP + i] = -1;
        }
        if (tid == 0) {
            ts[0] = 0;
            for (int e = 0; e < NEXP; ++e)
                ts[e + 1] = ts[e] + (cnt[e] + TILE_M - 1) / TILE_M;
        }
    }
}

// ---------------------------------------------------------------------------
// Kernel C1: pre-convert expert_w fp32 -> split bf16 (hi/lo), stored in the
// exact padded [n][KROW] tile layout the GEMM stages into LDS (so a B-tile is
// one contiguous 20 KB block -> single TDM descriptor). Done ONCE instead of
// once per consuming M-tile block (~64x VALU saved).
// ---------------------------------------------------------------------------
__global__ void __launch_bounds__(256)
moe_convw_kernel(const float* __restrict__ W,
                 unsigned short* __restrict__ whi, unsigned short* __restrict__ wlo) {
    const unsigned id  = blockIdx.x * 256 + threadIdx.x;   // 0 .. 8M-1
    const unsigned nc4 = id & 63;                          // n = nc4*4
    const unsigned nb  = (id >> 6) & 7;
    const unsigned kr  = (id >> 9) & 31;
    const unsigned kk  = (id >> 14) & 63;
    const unsigned e   = id >> 20;

    const f32x4 v = *(const f32x4*)(W + ((size_t)e * HDIM + kk * 32 + kr) * HDIM +
                                    nb * 256 + nc4 * 4);
    const size_t tile = ((size_t)e * KSTEPS + kk) * GRID_N + nb;
    size_t base = tile * (TILE_N * KROW) + (size_t)(nc4 * 4) * KROW + kr;
    unsigned short h, l;
    split_bf(v.x, h, l); whi[base]            = h; wlo[base]            = l;
    split_bf(v.y, h, l); whi[base + KROW]     = h; wlo[base + KROW]     = l;
    split_bf(v.z, h, l); whi[base + 2 * KROW] = h; wlo[base + 2 * KROW] = l;
    split_bf(v.w, h, l); whi[base + 3 * KROW] = h; wlo[base + 3 * KROW] = l;
}

// ---------------------------------------------------------------------------
// Kernel C2: pre-convert hidden fp32 -> split bf16 rows (row-major, so the
// GEMM's A staging is a TDM gather over 16-bit token indices).
// ---------------------------------------------------------------------------
__global__ void __launch_bounds__(256)
moe_convh_kernel(const float* __restrict__ hidden,
                 unsigned short* __restrict__ hhi, unsigned short* __restrict__ hlo) {
    const size_t off = ((size_t)blockIdx.x * 256 + threadIdx.x) * 4;
    const f32x4 v = *(const f32x4*)(hidden + off);
    u16x4 hv, lv;
    unsigned short h, l;
    split_bf(v.x, h, l); hv.x = h; lv.x = l;
    split_bf(v.y, h, l); hv.y = h; lv.y = l;
    split_bf(v.z, h, l); hv.z = h; lv.z = l;
    split_bf(v.w, h, l); hv.w = h; lv.w = l;
    *(u16x4*)(hhi + off) = hv;
    *(u16x4*)(hlo + off) = lv;
}

// ---------------------------------------------------------------------------
// Kernel 3 (fast path): routed expert GEMM, split-bf16 WMMA, TDM staging.
// All LDS staging is done by the Tensor Data Mover: 6 descriptors per k-step
// (B hi, B lo contiguous 20KB tiles; A hi/lo as 2x 16-row gathers with LDS
// padding to the 80B row stride), one descriptor issued per wave (scalar wid
// branch -- TDM ignores EXEC), double-buffered over TENSORcnt.
// ---------------------------------------------------------------------------
__global__ void __launch_bounds__(256)
moe_gemm_tdm_kernel(const unsigned short* __restrict__ hhi,
                    const unsigned short* __restrict__ hlo,
                    const unsigned short* __restrict__ whi,
                    const unsigned short* __restrict__ wlo,
                    const float* __restrict__ expert_b,
                    const int*   __restrict__ list,
                    const float* __restrict__ wslot,
                    const int*   __restrict__ tileStart,
                    float* __restrict__ out,
                    int accumulate) {
    __shared__ __align__(16) unsigned short As[2][2][TILE_M * KROW];  // [buf][hi/lo]
    __shared__ __align__(16) unsigned short Bs[2][2][TILE_N * KROW];  // [buf][hi/lo]
    __shared__ int   tok_sh[TILE_M];
    __shared__ float wgt_sh[TILE_M];
    __shared__ int   ts_sh[NEXP + 1];

    const int tid = threadIdx.x;
    if (tid <= NEXP) ts_sh[tid] = tileStart[tid];
    __syncthreads();

    const int my = blockIdx.y;
    if (my >= ts_sh[NEXP]) return;

    int e = 0;
    while (e < NEXP - 1 && my >= ts_sh[e + 1]) ++e;
    const int mt = my - ts_sh[e];

    if (tid < TILE_M) {
        const int t = list[e * CAP + mt * TILE_M + tid];
        tok_sh[tid] = t;
        wgt_sh[tid] = (t >= 0) ? wslot[t] : 0.0f;
    }
    __syncthreads();

    const int nbase = blockIdx.x * TILE_N;
    const int lane  = tid & 31;
    const int wave  = tid >> 5;
    const int wm    = wave >> 2;
    const int wn    = wave & 3;
    const int l15   = lane & 15;
    const int lhi   = lane >> 4;

    // ---- per-wave TDM descriptor setup (wid is an SGPR -> scalar branches) ----
    const int wid = rfl(tid) >> 5;
    unsigned long long gbase = 0, gstep = 0;
    unsigned ldsDst[2] = {0, 0};
    unsigned gather = 0;
    i32x8 g1 = {0, 0, 0, 0, 0, 0, 0, 0};
    i32x4 g2 = {0, 0, 0, 0};
    i32x4 g3 = {0, 0, 0, 0};

    if (wid < 2) {
        // B tile: contiguous 20480B (2560 x 8B elements), 1-D copy.
        const unsigned short* src = (wid == 0) ? whi : wlo;
        const size_t tile0 = ((size_t)e * KSTEPS) * GRID_N + blockIdx.x;
        gbase = (unsigned long long)(size_t)src + tile0 * W_TILE_BYTES;
        gstep = (unsigned long long)GRID_N * W_TILE_BYTES;
        ldsDst[0] = (unsigned)(size_t)&Bs[0][wid][0];
        ldsDst[1] = (unsigned)(size_t)&Bs[1][wid][0];
        g1 = tdm_group1(/*8B*/3, 0, 0, 0, /*tdim0*/2560, /*tdim1*/1,
                        /*tile0*/2560, /*tile1*/0, /*stride0*/2560);
    } else if (wid < 6) {
        // A gather: 16 rows of 16 dwords (64B) each, LDS-padded +4 dwords -> 80B.
        const int sel  = (wid - 2) >> 1;          // 0 = hi, 1 = lo
        const int half = (wid - 2) & 1;           // rows 0-15 / 16-31
        const unsigned short* src = sel ? hlo : hhi;
        gbase = (unsigned long long)(size_t)src;  // +kk*64B per step (column k0)
        gstep = TILE_K * 2;
        ldsDst[0] = (unsigned)(size_t)&As[0][sel][half * 16 * KROW];
        ldsDst[1] = (unsigned)(size_t)&As[1][sel][half * 16 * KROW];
        gather = 1;
        g1 = tdm_group1(/*4B*/2, /*pad*/1, /*16 dw*/3, /*4 dw*/3,
                        /*tdim0*/HDIM / 2, /*tdim1*/NTOK,
                        /*tile0*/16, /*#idx*/16, /*stride0*/HDIM / 2);
        const int base = half * 16;
#pragma unroll
        for (int j = 0; j < 4; ++j) {
            int a = tok_sh[base + 2 * j];
            int b = tok_sh[base + 2 * j + 1];
            a = (a < 0) ? 0 : a;                  // sentinel rows gather row 0
            b = (b < 0) ? 0 : b;                  // (result discarded, wgt==0)
            g2[j] = rfl((a & 0xFFFF) | (b << 16));
            a = tok_sh[base + 8 + 2 * j];
            b = tok_sh[base + 8 + 2 * j + 1];
            a = (a < 0) ? 0 : a;
            b = (b < 0) ? 0 : b;
            g3[j] = rfl((a & 0xFFFF) | (b << 16));
        }
    }

    v8f acc[4];
    const v8f vzero = {0.f, 0.f, 0.f, 0.f, 0.f, 0.f, 0.f, 0.f};
#pragma unroll
    for (int s = 0; s < 4; ++s) acc[s] = vzero;

    // ---- double-buffered TDM pipeline ----
    if (wid < 6) tdm_load(gather, ldsDst[0], gbase, g1, g2, g3);

    for (int kk = 0; kk < KSTEPS; ++kk) {
        __builtin_amdgcn_s_wait_tensorcnt(0);   // own TDM ops complete
        __syncthreads();                        // buf[kk&1] ready; prev reads done
        const int kn = kk + 1;
        if (kn < KSTEPS && wid < 6)
            tdm_load(gather, ldsDst[kn & 1], gbase + (unsigned long long)kn * gstep,
                     g1, g2, g3);

        const int buf = kk & 1;
        const unsigned short* Ah = As[buf][0];
        const unsigned short* Al = As[buf][1];
        const unsigned short* Bh = Bs[buf][0];
        const unsigned short* Bl = Bs[buf][1];

        const int arow = wm * 16 + l15;
        const int kba  = lhi ? 16 : 0;
        FragBF fa_hi, fa_lo;
        fa_hi.u[0] = *(const u32x4*)(Ah + arow * KROW + kba);
        fa_hi.u[1] = *(const u32x4*)(Ah + arow * KROW + kba + 8);
        fa_lo.u[0] = *(const u32x4*)(Al + arow * KROW + kba);
        fa_lo.u[1] = *(const u32x4*)(Al + arow * KROW + kba + 8);

        const int kbb = lhi ? 16 : 0;
#pragma unroll
        for (int s = 0; s < 4; ++s) {
            const int nloc = wn * 64 + s * 16 + l15;
            FragBF fb_hi, fb_lo;
            fb_hi.u[0] = *(const u32x4*)(Bh + nloc * KROW + kbb);
            fb_hi.u[1] = *(const u32x4*)(Bh + nloc * KROW + kbb + 8);
            fb_lo.u[0] = *(const u32x4*)(Bl + nloc * KROW + kbb);
            fb_lo.u[1] = *(const u32x4*)(Bl + nloc * KROW + kbb + 8);

            acc[s] = wmma_bf16(fa_hi.v, fb_hi.v, acc[s]);   // hi*hi
            acc[s] = wmma_bf16(fa_hi.v, fb_lo.v, acc[s]);   // hi*lo
            acc[s] = wmma_bf16(fa_lo.v, fb_hi.v, acc[s]);   // lo*hi
        }
    }

    const float* bias = expert_b + (size_t)e * HDIM;
#pragma unroll
    for (int s = 0; s < 4; ++s) {
        const int n = nbase + wn * 64 + s * 16 + l15;
        const float b = bias[n];
#pragma unroll
        for (int r = 0; r < 8; ++r) {
            const int mloc = wm * 16 + r + lhi * 8;
            const int t = tok_sh[mloc];
            if (t >= 0) {
                const float v = wgt_sh[mloc] * (acc[s][r] + b);
                float* p = out + (size_t)t * HDIM + n;
                if (accumulate) *p = *p + v;
                else            *p = v;
            }
        }
    }
}

// ---------------------------------------------------------------------------
// Kernel 3 (fallback, small-workspace): round-1 GEMM with in-block conversion.
// ---------------------------------------------------------------------------
__global__ void __launch_bounds__(256)
moe_gemm_fb_kernel(const float* __restrict__ hidden,
                   const float* __restrict__ expert_w,
                   const float* __restrict__ expert_b,
                   const int*   __restrict__ list,
                   const float* __restrict__ wslot,
                   const int*   __restrict__ tileStart,
                   float* __restrict__ out,
                   int accumulate) {
    __shared__ __align__(16) unsigned short As_hi[TILE_M][KROW];
    __shared__ __align__(16) unsigned short As_lo[TILE_M][KROW];
    __shared__ __align__(16) unsigned short Bt_hi[TILE_N][KROW];
    __shared__ __align__(16) unsigned short Bt_lo[TILE_N][KROW];
    __shared__ int   tok_sh[TILE_M];
    __shared__ float wgt_sh[TILE_M];
    __shared__ int   ts_sh[NEXP + 1];

    const int tid = threadIdx.x;
    if (tid <= NEXP) ts_sh[tid] = tileStart[tid];
    __syncthreads();

    const int my = blockIdx.y;
    if (my >= ts_sh[NEXP]) return;

    int e = 0;
    while (e < NEXP - 1 && my >= ts_sh[e + 1]) ++e;
    const int mt = my - ts_sh[e];

    if (tid < TILE_M) {
        const int t = list[e * CAP + mt * TILE_M + tid];
        tok_sh[tid] = t;
        wgt_sh[tid] = (t >= 0) ? wslot[t] : 0.0f;
    }
    __syncthreads();

    const int nbase = blockIdx.x * TILE_N;
    const int lane  = tid & 31;
    const int wave  = tid >> 5;
    const int wm    = wave >> 2;
    const int wn    = wave & 3;
    const int l15   = lane & 15;
    const int lhi   = lane >> 4;

    const float* wbase = expert_w + (size_t)e * HDIM * HDIM;

    v8f acc[4];
    const v8f vzero = {0.f, 0.f, 0.f, 0.f, 0.f, 0.f, 0.f, 0.f};
#pragma unroll
    for (int s = 0; s < 4; ++s) acc[s] = vzero;

    for (int kk = 0; kk < KSTEPS; ++kk) {
        const int k0 = kk * TILE_K;
        __syncthreads();
        {
            const int row = tid >> 3;
            const int ch  = (tid & 7) * 4;
            const int t   = tok_sh[row];
            f32x4 v = {0.f, 0.f, 0.f, 0.f};
            if (t >= 0) v = *(const f32x4*)(hidden + (size_t)t * HDIM + k0 + ch);
            unsigned short h, l;
            split_bf(v.x, h, l); As_hi[row][ch + 0] = h; As_lo[row][ch + 0] = l;
            split_bf(v.y, h, l); As_hi[row][ch + 1] = h; As_lo[row][ch + 1] = l;
            split_bf(v.z, h, l); As_hi[row][ch + 2] = h; As_lo[row][ch + 2] = l;
            split_bf(v.w, h, l); As_hi[row][ch + 3] = h; As_lo[row][ch + 3] = l;
        }
#pragma unroll
        for (int i = 0; i < 8; ++i) {
            const int id = i * 256 + tid;
            const int kr = id >> 6;
            const int nc = (id & 63) * 4;
            const f32x4 v =
                *(const f32x4*)(wbase + (size_t)(k0 + kr) * HDIM + nbase + nc);
            unsigned short h, l;
            split_bf(v.x, h, l); Bt_hi[nc + 0][kr] = h; Bt_lo[nc + 0][kr] = l;
            split_bf(v.y, h, l); Bt_hi[nc + 1][kr] = h; Bt_lo[nc + 1][kr] = l;
            split_bf(v.z, h, l); Bt_hi[nc + 2][kr] = h; Bt_lo[nc + 2][kr] = l;
            split_bf(v.w, h, l); Bt_hi[nc + 3][kr] = h; Bt_lo[nc + 3][kr] = l;
        }
        __syncthreads();

        const int arow = wm * 16 + l15;
        const int kba  = lhi ? 16 : 0;
        FragBF fa_hi, fa_lo;
        fa_hi.u[0] = *(const u32x4*)&As_hi[arow][kba];
        fa_hi.u[1] = *(const u32x4*)&As_hi[arow][kba + 8];
        fa_lo.u[0] = *(const u32x4*)&As_lo[arow][kba];
        fa_lo.u[1] = *(const u32x4*)&As_lo[arow][kba + 8];

        const int kbb = lhi ? 16 : 0;
#pragma unroll
        for (int s = 0; s < 4; ++s) {
            const int nloc = wn * 64 + s * 16 + l15;
            FragBF fb_hi, fb_lo;
            fb_hi.u[0] = *(const u32x4*)&Bt_hi[nloc][kbb];
            fb_hi.u[1] = *(const u32x4*)&Bt_hi[nloc][kbb + 8];
            fb_lo.u[0] = *(const u32x4*)&Bt_lo[nloc][kbb];
            fb_lo.u[1] = *(const u32x4*)&Bt_lo[nloc][kbb + 8];

            acc[s] = wmma_bf16(fa_hi.v, fb_hi.v, acc[s]);
            acc[s] = wmma_bf16(fa_hi.v, fb_lo.v, acc[s]);
            acc[s] = wmma_bf16(fa_lo.v, fb_hi.v, acc[s]);
        }
    }

    const float* bias = expert_b + (size_t)e * HDIM;
#pragma unroll
    for (int s = 0; s < 4; ++s) {
        const int n = nbase + wn * 64 + s * 16 + l15;
        const float b = bias[n];
#pragma unroll
        for (int r = 0; r < 8; ++r) {
            const int mloc = wm * 16 + r + lhi * 8;
            const int t = tok_sh[mloc];
            if (t >= 0) {
                const float v = wgt_sh[mloc] * (acc[s][r] + b);
                float* p = out + (size_t)t * HDIM + n;
                if (accumulate) *p = *p + v;
                else            *p = v;
            }
        }
    }
}

// ---------------------------------------------------------------------------
// Workspace: [0,1MB) routing metadata; then (fast path only)
// whi/wlo (80MB each, tiled) and hhi/hlo (32MB each, row-major).
// ---------------------------------------------------------------------------
extern "C" void kernel_launch(void* const* d_in, const int* in_sizes, int n_in,
                              void* d_out, int out_size, void* d_ws, size_t ws_size,
                              hipStream_t stream) {
    const float* hidden   = (const float*)d_in[0];
    const float* expert_w = (const float*)d_in[1];
    const float* expert_b = (const float*)d_in[2];
    const float* gate_w   = (const float*)d_in[3];
    const float* gate_b   = (const float*)d_in[4];
    float* out = (float*)d_out;

    int*   wsi = (int*)d_ws;
    float* wsf = (float*)d_ws;
    int*   cnt0  = wsi;
    int*   cnt1  = wsi + 8;
    int*   ts0   = wsi + 16;
    int*   ts1   = wsi + 32;
    float* w0    = wsf + 64;
    float* w1    = wsf + 64 + NTOK;
    int*   list0 = wsi + 64 + 2 * NTOK;
    int*   list1 = list0 + NEXP * CAP;

    moe_init_kernel<<<1, 32, 0, stream>>>(cnt0);
    moe_gate_kernel<<<NTOK, 256, 0, stream>>>(hidden, gate_w, gate_b,
                                              cnt0, cnt1, list0, list1, w0, w1);
    moe_finalize_kernel<<<1, 64, 0, stream>>>(cnt0, cnt1, list0, list1, ts0, ts1);

    dim3 grid(GRID_N, MAX_TILES);
    if (ws_size >= REQ_BYTES) {
        char* wsb = (char*)d_ws;
        unsigned short* whi = (unsigned short*)(wsb + META_BYTES);
        unsigned short* wlo = (unsigned short*)(wsb + META_BYTES + WPART_BYTES);
        unsigned short* hhi = (unsigned short*)(wsb + META_BYTES + 2 * WPART_BYTES);
        unsigned short* hlo = (unsigned short*)(wsb + META_BYTES + 2 * WPART_BYTES +
                                                HPART_BYTES);
        moe_convw_kernel<<<32768, 256, 0, stream>>>(expert_w, whi, wlo);
        moe_convh_kernel<<<16384, 256, 0, stream>>>(hidden, hhi, hlo);
        moe_gemm_tdm_kernel<<<grid, 256, 0, stream>>>(hhi, hlo, whi, wlo, expert_b,
                                                      list0, w0, ts0, out, 0);
        moe_gemm_tdm_kernel<<<grid, 256, 0, stream>>>(hhi, hlo, whi, wlo, expert_b,
                                                      list1, w1, ts1, out, 1);
    } else {
        moe_gemm_fb_kernel<<<grid, 256, 0, stream>>>(hidden, expert_w, expert_b,
                                                     list0, w0, ts0, out, 0);
        moe_gemm_fb_kernel<<<grid, 256, 0, stream>>>(hidden, expert_w, expert_b,
                                                     list1, w1, ts1, out, 1);
    }
}
